// CartesianTransformer_58626303590864
// MI455X (gfx1250) — matrix operand order repaired and checked by reading, not compiled, not run on hardware
//
#include <hip/hip_runtime.h>
#include <hip/hip_bf16.h>

typedef __attribute__((ext_vector_type(16))) _Float16 v16h;
typedef __attribute__((ext_vector_type(8)))  float    v8f;

#define DEVFN __device__ __forceinline__

static constexpr int Bsz  = 2048;
static constexpr int Ntok = 64;
static constexpr int Dm   = 256;
static constexpr int Sreal= 65;
static constexpr int Sp   = 80;    // padded sequence (5 x 16)
static constexpr int Hh   = 8;
static constexpr int DHd  = 32;
static constexpr int Lyr  = 3;
static constexpr int DFFd = 1024;

// f16 workspace weight offsets (in halves), all stored transposed: wT[n][k]
static constexpr size_t OFF_W1T  = 0;                          // 256 x 768
static constexpr size_t OFF_W2T  = OFF_W1T + 256*768;          // 256 x 256
static constexpr size_t OFF_QKVT = OFF_W2T + 256*256;          // 3 x 768 x 256
static constexpr size_t OFF_WOT  = OFF_QKVT + 3*768*256;       // 3 x 256 x 256
static constexpr size_t OFF_F1T  = OFF_WOT + 3*256*256;        // 3 x 1024 x 256
static constexpr size_t OFF_F2T  = OFF_F1T + 3*1024*256;       // 3 x 256 x 1024

// LDS layout (bytes)
static constexpr int SMO_POOL  = Sp*Dm*4;                  // x f32 80x256 = 81920
static constexpr int POOL_QKH  = 0;                        // 80x512 f16 = 81920
static constexpr int POOL_VT   = 81920;                    // 256x96 f16 = 49152
static constexpr int POOL_PH   = 131072;                   // 16*16*96 f16 = 49152
static constexpr int POOL_OH   = 180224;                   // 80x256 f16 = 40960
static constexpr int POOL_BYTES= 221184;
static constexpr int SMO_BIAS  = SMO_POOL + POOL_BYTES;    // 303104
static constexpr int SM_TOTAL  = SMO_BIAS + 512;           // 303616 <= 327680

DEVFN v8f zero8() { v8f z = {0.f,0.f,0.f,0.f,0.f,0.f,0.f,0.f}; return z; }

DEVFN float fast_rcp(float x) { return __builtin_amdgcn_rcpf(x); }
DEVFN float silu(float v) { return v * fast_rcp(1.f + __expf(-v)); }

DEVFN v8f wmma_f16(v16h a, v16h b, v8f c) {
  return __builtin_amdgcn_wmma_f32_16x16x32_f16(false, a, false, b, (short)0, c, false, false);
}

// A fragment 16x32 f16 from row-major f16 (stride ld halves), origin (row0,k0).
DEVFN v16h frag_a_f16(const _Float16* p, int ld, int row0, int k0) {
  const int lane = threadIdx.x & 31;
  const _Float16* q = p + (row0 + (lane & 15)) * ld + k0 + ((lane >> 4) << 3);
  v16h a;
#pragma unroll
  for (int v = 0; v < 8; ++v) {
    int k = ((v & 4) << 2) + ((v & 3) << 1);
    a[2*v]   = q[k];
    a[2*v+1] = q[k+1];
  }
  return a;
}

// A fragment from f32 source with convert-to-f16
DEVFN v16h frag_a_f32(const float* p, int ld, int row0, int k0) {
  const int lane = threadIdx.x & 31;
  const float* q = p + (row0 + (lane & 15)) * ld + k0 + ((lane >> 4) << 3);
  v16h a;
#pragma unroll
  for (int v = 0; v < 8; ++v) {
    int k = ((v & 4) << 2) + ((v & 3) << 1);
    a[2*v]   = (_Float16)q[k];
    a[2*v+1] = (_Float16)q[k+1];
  }
  return a;
}

// B fragment 32x16 from N-major storage src[n][k] (stride ldk halves): two 16B chunks (k=8h..8h+7, 16+8h..16+8h+7).
DEVFN v16h frag_b_nmajor(const _Float16* p, int ldk, int n0, int k0) {
  const int lane = threadIdx.x & 31;
  const _Float16* q = p + (size_t)(n0 + (lane & 15)) * ldk + k0 + ((lane >> 4) << 3);
  v16h b;
#pragma unroll
  for (int s = 0; s < 8; ++s) {
    b[s]     = q[s];
    b[8 + s] = q[16 + s];
  }
  return b;
}

// C/D layout: lane col n=n0+(lane&15); VGPR r row = row0 + 8*(lane>=16) + r
DEVFN v8f load_c_f32(const float* p, int ld, int row0, int n0) {
  const int lane = threadIdx.x & 31;
  const float* q = p + (size_t)(row0 + ((lane >> 4) << 3)) * ld + n0 + (lane & 15);
  v8f c;
#pragma unroll
  for (int r = 0; r < 8; ++r) c[r] = q[(size_t)r * ld];
  return c;
}
DEVFN void store_c_f32(float* p, int ld, int row0, int n0, v8f c) {
  const int lane = threadIdx.x & 31;
  float* q = p + (size_t)(row0 + ((lane >> 4) << 3)) * ld + n0 + (lane & 15);
#pragma unroll
  for (int r = 0; r < 8; ++r) q[(size_t)r * ld] = c[r];
}

DEVFN void ln_row(const float* y, float* xo, const float* g, const float* be, int row) {
  const int lane = threadIdx.x & 31;
  const float* yr = y + (size_t)row * Dm;
  float s = 0.f, s2 = 0.f;
#pragma unroll
  for (int i = 0; i < 8; ++i) { float v = yr[lane + i*32]; s += v; s2 += v*v; }
#pragma unroll
  for (int m = 1; m < 32; m <<= 1) { s += __shfl_xor(s, m, 32); s2 += __shfl_xor(s2, m, 32); }
  float mean = s * (1.f/256.f);
  float var  = s2 * (1.f/256.f) - mean*mean;
  float inv  = __builtin_amdgcn_rsqf(var + 1e-5f);
  float* xr = xo + (size_t)row * Dm;
#pragma unroll
  for (int i = 0; i < 8; ++i) {
    int c = lane + i*32;
    xr[c] = (yr[c] - mean) * inv * g[c] + be[c];
  }
}

// Prefetch a weight span into caches (global_prefetch_b8), 1 byte per 256B line.
DEVFN void prefetch_span(const void* p, int bytes) {
  const char* c = (const char*)p;
  for (int off = (int)threadIdx.x * 256; off < bytes; off += 512 * 256)
    __builtin_prefetch(c + off, 0, 1);
}

// ---------------- weight transpose+cast: dst[n*K+k] = src[k*N+n] ----------------
__global__ void prep_transpose(const float* __restrict__ src, _Float16* __restrict__ dst,
                               int K, int N) {
  int i = blockIdx.x * blockDim.x + threadIdx.x;
  if (i >= K * N) return;
  int n = i / K, k = i - n * K;
  dst[i] = (_Float16)src[(size_t)k * N + n];
}

// ---------------- fused per-sequence transformer ----------------
__global__ __launch_bounds__(512, 1)
void fused_transformer(
    const float* __restrict__ input_messages, const float* __restrict__ momenta,
    const float* __restrict__ edge_vectors,   const float* __restrict__ edge_distances,
    const float* __restrict__ cutoff_factors, const unsigned char* __restrict__ padding_mask,
    const int* __restrict__ elem_nodes,       const int* __restrict__ elem_nbrs,
    const float* __restrict__ edge_w, const float* __restrict__ edge_b,
    const float* __restrict__ node_table,
    const float* __restrict__ mom_w, const float* __restrict__ mom_b,
    const float* __restrict__ node_comp_w, const float* __restrict__ node_comp_b,
    const float* __restrict__ nbr_table,
    const float* __restrict__ comp_b1, const float* __restrict__ comp_b2,
    const float* __restrict__ qkv_b_all, const float* __restrict__ attn_out_b_all,
    const float* __restrict__ ln1_g_all, const float* __restrict__ ln1_b_all,
    const float* __restrict__ ffn_b1_all, const float* __restrict__ ffn_b2_all,
    const float* __restrict__ ln2_g_all, const float* __restrict__ ln2_b_all,
    const _Float16* __restrict__ wsh,
    float* __restrict__ out) {
  extern __shared__ __align__(16) char smem[];
  float* xf      = (float*)smem;                 // 80 x 256 f32, rows 65..79 zero
  char*  pool    = smem + SMO_POOL;              // phase-reused region
  float* biasv   = (float*)(smem + SMO_BIAS);    // 80 f32 softmax bias (key axis)

  const int bi   = blockIdx.x;
  const int tid  = threadIdx.x;
  const int wv   = tid >> 5;
  const int lane = tid & 31;

  const _Float16* w1T = wsh + OFF_W1T;
  const _Float16* w2T = wsh + OFF_W2T;

  // ================= build phase =================
  {
    _Float16* cat = (_Float16*)pool;             // 64 x 768 f16
    _Float16* hb  = (_Float16*)(pool + 98304);   // 64 x 256 f16
    float*    mom = (float*)(pool + 131072);     // 256 f32

    for (int i = tid; i < Ntok*Dm; i += 512) {
      int r = i >> 8, c = i & 255;
      float e0 = edge_vectors[((size_t)bi*Ntok + r)*3 + 0];
      float e1 = edge_vectors[((size_t)bi*Ntok + r)*3 + 1];
      float e2 = edge_vectors[((size_t)bi*Ntok + r)*3 + 2];
      float dd = edge_distances[(size_t)bi*Ntok + r];
      float ev = e0*edge_w[c] + e1*edge_w[Dm+c] + e2*edge_w[2*Dm+c] + dd*edge_w[3*Dm+c] + edge_b[c];
      cat[r*768 + c]        = (_Float16)ev;
      int en = elem_nbrs[(size_t)bi*Ntok + r];
      cat[r*768 + Dm + c]   = (_Float16)nbr_table[(size_t)en*Dm + c];
      cat[r*768 + 2*Dm + c] = (_Float16)input_messages[((size_t)bi*Ntok + r)*Dm + c];
    }
    if (tid < Dm) {
      float m0 = momenta[bi*3], m1 = momenta[bi*3+1], m2 = momenta[bi*3+2];
      mom[tid] = m0*mom_w[tid] + m1*mom_w[Dm+tid] + m2*mom_w[2*Dm+tid] + mom_b[tid];
    }
    for (int s = tid; s < Sp; s += 512) {
      float bv;
      if (s == 0) bv = 0.f;
      else if (s < Sreal) {
        int j = s - 1;
        float cf = padding_mask[(size_t)bi*Ntok + j] ? cutoff_factors[(size_t)bi*Ntok + j] : 0.f;
        bv = (cf > 0.f) ? logf(fmaxf(cf, 1e-30f)) : -1.0e9f;
      } else bv = -1.0e9f;
      biasv[s] = bv;
    }
    for (int i = tid; i < (Sp - Sreal)*Dm; i += 512) xf[Sreal*Dm + i] = 0.f;
    prefetch_span(wsh + OFF_QKVT, 768*256*2);   // overlap layer-0 weight fill with build GEMMs
    __syncthreads();

    // h = silu(cat @ comp_w1 + b1)   M=64 N=256 K=768; wave owns N-tile, M-blocked x4
    {
      const int nt = wv;
      v8f acc[4];
#pragma unroll
      for (int mt = 0; mt < 4; ++mt) acc[mt] = zero8();
#pragma unroll 2
      for (int kt = 0; kt < 24; ++kt) {
        v16h bfr = frag_b_nmajor(w1T, 768, nt*16, kt*32);
#pragma unroll
        for (int mt = 0; mt < 4; ++mt)
          acc[mt] = wmma_f16(frag_a_f16(cat, 768, mt*16, kt*32), bfr, acc[mt]);
      }
      int n = nt*16 + (lane & 15);
      float b1v = comp_b1[n];
#pragma unroll
      for (int mt = 0; mt < 4; ++mt) {
        int mb = mt*16 + ((lane >> 4) << 3);
#pragma unroll
        for (int r = 0; r < 8; ++r)
          hb[(mb + r)*Dm + n] = (_Float16)silu(acc[mt][r] + b1v);
      }
    }
    __syncthreads();

    // tok = h @ comp_w2 + b2 -> xf rows 1..64   M=64 N=256 K=256
    {
      const int nt = wv;
      v8f acc[4];
#pragma unroll
      for (int mt = 0; mt < 4; ++mt) acc[mt] = zero8();
#pragma unroll 2
      for (int kt = 0; kt < 8; ++kt) {
        v16h bfr = frag_b_nmajor(w2T, Dm, nt*16, kt*32);
#pragma unroll
        for (int mt = 0; mt < 4; ++mt)
          acc[mt] = wmma_f16(frag_a_f16(hb, Dm, mt*16, kt*32), bfr, acc[mt]);
      }
      int n = nt*16 + (lane & 15);
      float b2v = comp_b2[n];
#pragma unroll
      for (int mt = 0; mt < 4; ++mt) {
        int mb = 1 + mt*16 + ((lane >> 4) << 3);
#pragma unroll
        for (int r = 0; r < 8; ++r) xf[(mb + r)*Dm + n] = acc[mt][r] + b2v;
      }
    }
    // node row 0: concat(node_table[idx], mom) @ node_comp_w + b
    if (tid < Dm) {
      int ni = elem_nodes[bi];
      float acc = node_comp_b[tid];
      const float* nrow = node_table + (size_t)ni*Dm;
      for (int k = 0; k < Dm; ++k)  acc += nrow[k] * node_comp_w[(size_t)k*Dm + tid];
      for (int k = 0; k < Dm; ++k)  acc += mom[k]  * node_comp_w[(size_t)(Dm + k)*Dm + tid];
      xf[tid] = acc;
    }
    __syncthreads();
  }

  // ================= transformer layers =================
  _Float16* qkh = (_Float16*)(pool + POOL_QKH);  // 80 x 512 f16 (Q cols 0..255, K cols 256..511)
  _Float16* vT  = (_Float16*)(pool + POOL_VT);   // 256 x 96 f16: vT[c][key], keys 80..95 zero
  _Float16* ph  = (_Float16*)(pool + POOL_PH);   // per-wave 16 x 96 f16 probs scratch
  _Float16* oh  = (_Float16*)(pool + POOL_OH);   // 80 x 256 f16 attention output

  for (int ly = 0; ly < Lyr; ++ly) {
    const _Float16* qkvT = wsh + OFF_QKVT + (size_t)ly*768*256;
    const _Float16* woT  = wsh + OFF_WOT  + (size_t)ly*256*256;
    const _Float16* f1T  = wsh + OFF_F1T  + (size_t)ly*1024*256;
    const _Float16* f2T  = wsh + OFF_F2T  + (size_t)ly*256*1024;
    const float* qb  = qkv_b_all     + (size_t)ly*768;
    const float* ob  = attn_out_b_all+ (size_t)ly*256;
    const float* g1  = ln1_g_all + (size_t)ly*256;
    const float* be1 = ln1_b_all + (size_t)ly*256;
    const float* fb1 = ffn_b1_all + (size_t)ly*1024;
    const float* fb2 = ffn_b2_all + (size_t)ly*256;
    const float* g2  = ln2_g_all + (size_t)ly*256;
    const float* be2 = ln2_b_all + (size_t)ly*256;

    // ---- QKV: M=80 N=768 K=256; wave owns N-tile (3 per wave), M-blocked x5 ----
    for (int nt = wv; nt < 48; nt += 16) {
      v8f acc[5];
#pragma unroll
      for (int mt = 0; mt < 5; ++mt) acc[mt] = zero8();
#pragma unroll 2
      for (int kt = 0; kt < 8; ++kt) {
        v16h bfr = frag_b_nmajor(qkvT, Dm, nt*16, kt*32);
#pragma unroll
        for (int mt = 0; mt < 5; ++mt)
          acc[mt] = wmma_f16(frag_a_f32(xf, Dm, mt*16, kt*32), bfr, acc[mt]);
      }
      int n = nt*16 + (lane & 15);
      float bv = qb[n];
      if (nt < 32) {
#pragma unroll
        for (int mt = 0; mt < 5; ++mt) {
          int mb = mt*16 + ((lane >> 4) << 3);
#pragma unroll
          for (int r = 0; r < 8; ++r) qkh[(mb + r)*512 + n] = (_Float16)(acc[mt][r] + bv);
        }
      } else {
        // V transposed: vT[c][key]; contiguous along key -> vector ds stores
        _Float16* vq = vT + (size_t)(n - 512)*96;
#pragma unroll
        for (int mt = 0; mt < 5; ++mt) {
          int mb = mt*16 + ((lane >> 4) << 3);
#pragma unroll
          for (int r = 0; r < 8; ++r) vq[mb + r] = (_Float16)(acc[mt][r] + bv);
        }
      }
    }
    // zero vT padded keys 80..95
    for (int i = tid; i < Dm*16; i += 512) {
      int c = i >> 4, k = 80 + (i & 15);
      vT[c*96 + k] = (_Float16)0.f;
    }
    prefetch_span(woT, 256*256*2);
    prefetch_span(f1T, 1024*256*2);
    __syncthreads();

    // ---- attention: 8 heads x 5 query row-blocks = 40 units ----
    {
      const float scl = 0.17677669529663687f; // 1/sqrt(32)
      for (int u = wv; u < Hh*5; u += 16) {
        int hh = u / 5, mt = u % 5;
        v16h aq = frag_a_f16(qkh + hh*DHd, 512, mt*16, 0);
        v8f sc[5];
#pragma unroll
        for (int nt = 0; nt < 5; ++nt)
          sc[nt] = wmma_f16(aq, frag_b_nmajor(qkh + Dm + hh*DHd, 512, nt*16, 0), zero8());
#pragma unroll
        for (int nt = 0; nt < 5; ++nt) {
          float bv = biasv[nt*16 + (lane & 15)];
#pragma unroll
          for (int r = 0; r < 8; ++r) sc[nt][r] = sc[nt][r] * scl + bv;
        }
        // softmax along keys; each row lives in one 16-lane half (n axis)
#pragma unroll
        for (int r = 0; r < 8; ++r) {
          float mx = sc[0][r];
#pragma unroll
          for (int nt = 1; nt < 5; ++nt) mx = fmaxf(mx, sc[nt][r]);
#pragma unroll
          for (int m = 1; m < 16; m <<= 1) mx = fmaxf(mx, __shfl_xor(mx, m, 32));
          float sum = 0.f;
#pragma unroll
          for (int nt = 0; nt < 5; ++nt) { float e = __expf(sc[nt][r] - mx); sc[nt][r] = e; sum += e; }
#pragma unroll
          for (int m = 1; m < 16; m <<= 1) sum += __shfl_xor(sum, m, 32);
          float inv = fast_rcp(sum);
#pragma unroll
          for (int nt = 0; nt < 5; ++nt) sc[nt][r] *= inv;
        }
        // C-layout -> A-layout via per-wave LDS scratch (16 x 96, cols 80..95 zero)
        _Float16* pw = ph + wv * (16*96);
        {
          int n = lane & 15, mb = (lane >> 4) << 3;
#pragma unroll
          for (int nt = 0; nt < 5; ++nt)
#pragma unroll
            for (int r = 0; r < 8; ++r)
              pw[(mb + r)*96 + nt*16 + n] = (_Float16)sc[nt][r];
#pragma unroll
          for (int r = 0; r < 8; ++r) pw[(mb + r)*96 + 80 + n] = (_Float16)0.f;
        }
        // o = P @ V_h   (16 x 32), K=96; V fragments contiguous from vT
        v8f o0 = zero8(), o1 = zero8();
        const _Float16* vh = vT + (size_t)(hh*DHd)*96;
#pragma unroll
        for (int kt = 0; kt < 3; ++kt) {
          v16h ap = frag_a_f16(pw, 96, 0, kt*32);
          o0 = wmma_f16(ap, frag_b_nmajor(vh, 96, 0,  kt*32), o0);
          o1 = wmma_f16(ap, frag_b_nmajor(vh, 96, 16, kt*32), o1);
        }
        {
          int n = lane & 15, mb = mt*16 + ((lane >> 4) << 3);
#pragma unroll
          for (int r = 0; r < 8; ++r) {
            oh[(mb + r)*Dm + hh*DHd + n]      = (_Float16)o0[r];
            oh[(mb + r)*Dm + hh*DHd + 16 + n] = (_Float16)o1[r];
          }
        }
      }
      __syncthreads();
    }

    // ---- attn_out + residual -> yf ; LN1 -> xf ----
    {
      float* yf = (float*)pool;   // 80 x 256 f32 (overlays freed qkh)
      const int nt = wv;
      v8f acc[5];
#pragma unroll
      for (int mt = 0; mt < 5; ++mt) acc[mt] = zero8();
#pragma unroll 2
      for (int kt = 0; kt < 8; ++kt) {
        v16h bfr = frag_b_nmajor(woT, Dm, nt*16, kt*32);
#pragma unroll
        for (int mt = 0; mt < 5; ++mt)
          acc[mt] = wmma_f16(frag_a_f16(oh, Dm, mt*16, kt*32), bfr, acc[mt]);
      }
      int n = nt*16 + (lane & 15);
      float bv = ob[n];
#pragma unroll
      for (int mt = 0; mt < 5; ++mt) {
        int mb = mt*16 + ((lane >> 4) << 3);
#pragma unroll
        for (int r = 0; r < 8; ++r)
          yf[(mb + r)*Dm + n] = acc[mt][r] + bv + xf[(mb + r)*Dm + n];
      }
      __syncthreads();
      for (int r = wv; r < Sreal; r += 16) ln_row(yf, xf, g1, be1, r);
      __syncthreads();
    }

    // ---- FFN in two DFF halves ----
    {
      _Float16* h1 = (_Float16*)pool;           // 80 x 512 f16
      float*    y2 = (float*)(pool + 81920);    // 80 x 256 f32 accumulator
      for (int i = tid; i < Sp*Dm; i += 512) y2[i] = xf[i] + fb2[i & 255];
      prefetch_span(f2T, 1024*256*2);
      if (ly + 1 < Lyr) prefetch_span(wsh + OFF_QKVT + (size_t)(ly+1)*768*256, 768*256*2);
      __syncthreads();
      for (int half = 0; half < 2; ++half) {
        const _Float16* w1h = f1T + (size_t)half*512*256;
        for (int nt = wv; nt < 32; nt += 16) {  // M=80 N=512 K=256
          v8f acc[5];
#pragma unroll
          for (int mt = 0; mt < 5; ++mt) acc[mt] = zero8();
#pragma unroll 2
          for (int kt = 0; kt < 8; ++kt) {
            v16h bfr = frag_b_nmajor(w1h, Dm, nt*16, kt*32);
#pragma unroll
            for (int mt = 0; mt < 5; ++mt)
              acc[mt] = wmma_f16(frag_a_f32(xf, Dm, mt*16, kt*32), bfr, acc[mt]);
          }
          int n = nt*16 + (lane & 15);
          float bv = fb1[half*512 + n];
#pragma unroll
          for (int mt = 0; mt < 5; ++mt) {
            int mb = mt*16 + ((lane >> 4) << 3);
#pragma unroll
            for (int r = 0; r < 8; ++r)
              h1[(mb + r)*512 + n] = (_Float16)silu(acc[mt][r] + bv);
          }
        }
        __syncthreads();
        {                                       // y2 += h1 @ W2[half]  M=80 N=256 K=512
          const int nt = wv;
          v8f acc[5];
#pragma unroll
          for (int mt = 0; mt < 5; ++mt) acc[mt] = load_c_f32(y2, Dm, mt*16, nt*16);
#pragma unroll 2
          for (int kt = 0; kt < 16; ++kt) {
            v16h bfr = frag_b_nmajor(f2T, DFFd, nt*16, half*512 + kt*32);
#pragma unroll
            for (int mt = 0; mt < 5; ++mt)
              acc[mt] = wmma_f16(frag_a_f16(h1, 512, mt*16, kt*32), bfr, acc[mt]);
          }
#pragma unroll
          for (int mt = 0; mt < 5; ++mt) store_c_f32(y2, Dm, mt*16, nt*16, acc[mt]);
        }
        __syncthreads();
      }
      for (int r = wv; r < Sreal; r += 16) ln_row(y2, xf, g2, be2, r);
      __syncthreads();
    }
  }

  // ================= output: (x[:,0,:], x[:,1:,:]) concatenated =================
  {
    float* out0 = out;
    float* out1 = out + (size_t)Bsz*Dm;
    if (tid < Dm) out0[(size_t)bi*Dm + tid] = xf[tid];
    for (int i = tid; i < Ntok*Dm; i += 512) out1[(size_t)bi*Ntok*Dm + i] = xf[Dm + i];
  }
}

extern "C" void kernel_launch(void* const* d_in, const int* in_sizes, int n_in,
                              void* d_out, int out_size, void* d_ws, size_t ws_size,
                              hipStream_t stream) {
  (void)in_sizes; (void)n_in; (void)out_size; (void)ws_size;
  const float* input_messages = (const float*)d_in[0];
  const float* momenta        = (const float*)d_in[1];
  const float* edge_vectors   = (const float*)d_in[2];
  const float* edge_distances = (const float*)d_in[3];
  const float* cutoff_factors = (const float*)d_in[4];
  const unsigned char* padding_mask = (const unsigned char*)d_in[5];
  const int* elem_nodes = (const int*)d_in[6];
  const int* elem_nbrs  = (const int*)d_in[7];
  const float* edge_w      = (const float*)d_in[8];
  const float* edge_b      = (const float*)d_in[9];
  const float* node_table  = (const float*)d_in[10];
  const float* mom_w       = (const float*)d_in[11];
  const float* mom_b       = (const float*)d_in[12];
  const float* node_comp_w = (const float*)d_in[13];
  const float* node_comp_b = (const float*)d_in[14];
  const float* nbr_table   = (const float*)d_in[15];
  const float* comp_w1     = (const float*)d_in[16];
  const float* comp_b1     = (const float*)d_in[17];
  const float* comp_w2     = (const float*)d_in[18];
  const float* comp_b2     = (const float*)d_in[19];
  const float* qkv_w       = (const float*)d_in[20];
  const float* qkv_b       = (const float*)d_in[21];
  const float* attn_out_w  = (const float*)d_in[22];
  const float* attn_out_b  = (const float*)d_in[23];
  const float* ln1_g       = (const float*)d_in[24];
  const float* ln1_b       = (const float*)d_in[25];
  const float* ffn_w1      = (const float*)d_in[26];
  const float* ffn_b1      = (const float*)d_in[27];
  const float* ffn_w2      = (const float*)d_in[28];
  const float* ffn_b2      = (const float*)d_in[29];
  const float* ln2_g       = (const float*)d_in[30];
  const float* ln2_b       = (const float*)d_in[31];

  _Float16* wsh = (_Float16*)d_ws;
  auto tp = [&](const float* s, _Float16* d, int K, int N) {
    int n = K * N;
    prep_transpose<<<dim3((n + 255) / 256), dim3(256), 0, stream>>>(s, d, K, N);
  };
  tp(comp_w1, wsh + OFF_W1T, 768, 256);
  tp(comp_w2, wsh + OFF_W2T, 256, 256);
  for (int ly = 0; ly < 3; ++ly) {
    tp(qkv_w      + (size_t)ly*256*768,  wsh + OFF_QKVT + (size_t)ly*768*256,  256, 768);
    tp(attn_out_w + (size_t)ly*256*256,  wsh + OFF_WOT  + (size_t)ly*256*256,  256, 256);
    tp(ffn_w1     + (size_t)ly*256*1024, wsh + OFF_F1T  + (size_t)ly*1024*256, 256, 1024);
    tp(ffn_w2     + (size_t)ly*1024*256, wsh + OFF_F2T  + (size_t)ly*256*1024, 1024, 256);
  }

  fused_transformer<<<dim3(Bsz), dim3(512), SM_TOTAL, stream>>>(
      input_messages, momenta, edge_vectors, edge_distances, cutoff_factors,
      padding_mask, elem_nodes, elem_nbrs,
      edge_w, edge_b, node_table, mom_w, mom_b, node_comp_w, node_comp_b, nbr_table,
      comp_b1, comp_b2, qkv_b, attn_out_b, ln1_g, ln1_b, ffn_b1, ffn_b2, ln2_g, ln2_b,
      wsh, (float*)d_out);
}
